// MambaSeq2Seq_83227876262104
// MI455X (gfx1250) — matrix operand order repaired and checked by reading, not compiled, not run on hardware
//
#include <hip/hip_runtime.h>

typedef __attribute__((ext_vector_type(16))) _Float16 v16h;
typedef __attribute__((ext_vector_type(8)))  _Float16 v8h;
typedef __attribute__((ext_vector_type(8)))  float    v8f;

#define B_   4
#define L_   1024
#define DM   256
#define DI   512
#define DS   16
#define NL   4
#define M_   (B_ * L_)   // 4096 rows (tokens)

__device__ __forceinline__ float silu_f(float x) { return x / (1.0f + expf(-x)); }
__device__ __forceinline__ float softplus_f(float x) {
    return (x > 20.0f) ? x : log1pf(expf(x));
}

// ---------------------------------------------------------------------------
// Weight transpose + f32 -> f16 convert: W (N,K) row-major -> Wt (Kpad,N) f16,
// rows K..Kpad-1 zero-filled (lets the GEMM run an unguarded K-loop).
// ---------------------------------------------------------------------------
__global__ void k_transpose_f16(const float* __restrict__ W, _Float16* __restrict__ Wt,
                                int N, int K, int Kpad) {
    int t = blockIdx.x * blockDim.x + threadIdx.x;
    if (t >= N * Kpad) return;
    int k = t / N, n = t % N;
    Wt[t] = (k < K) ? (_Float16)W[(size_t)n * K + k] : (_Float16)0.0f;
}

// ---------------------------------------------------------------------------
// Embedding conv (k=3, SAME) + BatchNorm + SiLU, output (B*L, DM) f32 + f16.
// ---------------------------------------------------------------------------
__global__ void k_embed(const float* __restrict__ x, const float* __restrict__ ew,
                        const float* __restrict__ gam, const float* __restrict__ bet,
                        const float* __restrict__ mea, const float* __restrict__ var,
                        float* __restrict__ h, _Float16* __restrict__ hh) {
    int t = blockIdx.x * blockDim.x + threadIdx.x;
    if (t >= M_ * DM) return;
    int c = t & (DM - 1);
    int m = t >> 8;                 // b*L + l
    int b = m >> 10, l = m & (L_ - 1);
    const float* xr = x + (size_t)b * L_;
    float acc = 0.0f;
#pragma unroll
    for (int j = 0; j < 3; ++j) {
        int tt = l - 1 + j;
        float xv = (tt >= 0 && tt < L_) ? xr[tt] : 0.0f;
        acc += xv * ew[c * 3 + j];
    }
    float inv = rsqrtf(var[c] + 1e-5f);
    float v = (acc - mea[c]) * (inv * gam[c]) + bet[c];
    v = silu_f(v);
    h[t] = v;
    hh[t] = (_Float16)v;
}

// ---------------------------------------------------------------------------
// Register-blocked WMMA GEMM: C[m][n] = sum_k A[m][k] * Wt[k][n]
// Each wave owns a (16*BM) x (16*BN) macro-tile: per K-step, 2*BM b128 A loads + 16*BN f16 B loads
// feed BM*BN v_wmma_f32_16x16x32_f16.
//   A  : f16, row-major, lda (lda*2 bytes multiple of 16)
//   Wt : f16 pre-transposed weight, zero-padded so K % 32 == 0, row stride ldb
//   C  : f32, ldc
// mode 0: store f32   mode 1: softplus(c + bias[n]) f32   mode 2: f32 + f16
// ---------------------------------------------------------------------------
template <int BM, int BN>
__global__ void k_wmma_gemm(const _Float16* __restrict__ A, const _Float16* __restrict__ Wt,
                            float* __restrict__ C, _Float16* __restrict__ Ch,
                            const float* __restrict__ bias,
                            int M, int N, int K, int lda, int ldb, int ldc, int mode) {
    int wavesPerBlk = blockDim.x >> 5;
    int wave = blockIdx.x * wavesPerBlk + (threadIdx.x >> 5);
    int mtM = M / (16 * BM), mtN = N / (16 * BN);
    if (wave >= mtM * mtN) return;           // wave-uniform: EXEC stays all-ones
    int tm = wave / mtN, tn = wave % mtN;

    int lane  = threadIdx.x & 31;
    int khalf = (lane >> 4) << 3;            // 0 or 8 (A-fragment half-K split)

    v8f acc[BM][BN] = {};                    // zero-initialized accumulators

    const _Float16* arow[BM];
#pragma unroll
    for (int i = 0; i < BM; ++i)
        arow[i] = A + (size_t)(tm * BM * 16 + i * 16 + (lane & 15)) * lda + khalf;
    const _Float16* brow[BN];
#pragma unroll
    for (int j = 0; j < BN; ++j)
        brow[j] = Wt + (size_t)khalf * ldb + tn * BN * 16 + j * 16 + (lane & 15);

#pragma unroll 2
    for (int k0 = 0; k0 < K; k0 += 32) {
        __builtin_prefetch(arow[0] + k0 + 64, 0, 1);   // speculative prefetch
        v16h afrag[BM];
#pragma unroll
        for (int i = 0; i < BM; ++i) {
            // A fragment 16x32 (ISA 7.12.2): VGPRs 0..3 -> K=khalf+0..7,
            // VGPRs 4..7 -> K=16+khalf+0..7  => two b128 loads.
            v8h alo = *(const v8h*)(arow[i] + k0);
            v8h ahi = *(const v8h*)(arow[i] + k0 + 16);
            afrag[i] = __builtin_shufflevector(alo, ahi, 0, 1, 2, 3, 4, 5, 6, 7,
                                                         8, 9, 10, 11, 12, 13, 14, 15);
        }
        v16h bfrag[BN];
#pragma unroll
        for (int j = 0; j < BN; ++j) {
            // B fragment 32x16: lane l holds column N=l&15, K order as A fragment.
            const _Float16* wrow = brow[j] + (size_t)k0 * ldb;
#pragma unroll
            for (int s = 0; s < 8; ++s) {
                bfrag[j][s]     = wrow[(size_t)s * ldb];
                bfrag[j][8 + s] = wrow[(size_t)(16 + s) * ldb];
            }
        }
#pragma unroll
        for (int i = 0; i < BM; ++i)
#pragma unroll
            for (int j = 0; j < BN; ++j)
                acc[i][j] = __builtin_amdgcn_wmma_f32_16x16x32_f16(
                    false, afrag[i], false, bfrag[j],
                    (short)0, acc[i][j], false, false);
    }

    // C/D layout: VGPR r, lanes 0-15 -> M=r, lanes 16-31 -> M=r+8; N=lane&15.
    int rbase = (lane >> 4) << 3;
    int colb  = lane & 15;
#pragma unroll
    for (int i = 0; i < BM; ++i) {
#pragma unroll
        for (int j = 0; j < BN; ++j) {
            int col = tn * BN * 16 + j * 16 + colb;
#pragma unroll
            for (int r = 0; r < 8; ++r) {
                int row = tm * BM * 16 + i * 16 + rbase + r;
                float v = acc[i][j][r];
                if (mode == 1) v = softplus_f(v + bias[col]);
                C[(size_t)row * ldc + col] = v;
                if (mode == 2) Ch[(size_t)row * ldc + col] = (_Float16)v;
            }
        }
    }
}

// ---------------------------------------------------------------------------
// Depthwise causal conv (k=4, left pad 3) + bias + SiLU on xi, and silu(z).
// xz: (B*L, 2*DI) f32.  Writes xc f32, xc f16, zs f32 (all (B*L, DI)).
// ---------------------------------------------------------------------------
__global__ void k_conv_silu(const float* __restrict__ xz, const float* __restrict__ cw,
                            const float* __restrict__ cb,
                            float* __restrict__ xc, _Float16* __restrict__ xch,
                            float* __restrict__ zs) {
    int t = blockIdx.x * blockDim.x + threadIdx.x;
    if (t >= M_ * DI) return;
    int d = t & (DI - 1);
    int m = t >> 9;
    int b = m >> 10, l = m & (L_ - 1);
    float acc = cb[d];
#pragma unroll
    for (int j = 0; j < 4; ++j) {
        int tt = l - 3 + j;
        float xv = (tt >= 0) ? xz[((size_t)(b * L_ + tt)) * (2 * DI) + d] : 0.0f;
        acc += xv * cw[d * 4 + j];
    }
    float v = silu_f(acc);
    xc[t]  = v;
    xch[t] = (_Float16)v;
    zs[t]  = silu_f(xz[(size_t)m * (2 * DI) + DI + d]);
}

// ---------------------------------------------------------------------------
// Selective scan: one lane per (b, d). 16 states held in VGPRs, sequential
// over L. Output y = (scan + u*D) * silu(z), stored f16 for out_proj WMMA.
// ---------------------------------------------------------------------------
__global__ void k_scan(const float* __restrict__ dt, const float* __restrict__ xc,
                       const float* __restrict__ xdbl, const float* __restrict__ zs,
                       const float* __restrict__ A_log, const float* __restrict__ Dsk,
                       _Float16* __restrict__ yh) {
    int t = blockIdx.x * blockDim.x + threadIdx.x;
    if (t >= B_ * DI) return;
    int b = t / DI, d = t % DI;
    float Av[DS], hs[DS];
#pragma unroll
    for (int n = 0; n < DS; ++n) {
        Av[n] = -expf(A_log[d * DS + n]);
        hs[n] = 0.0f;
    }
    float Dv = Dsk[d];
    for (int l = 0; l < L_; ++l) {
        size_t m = (size_t)b * L_ + l;
        float dtv = dt[m * DI + d];
        float u   = xc[m * DI + d];
        const float* xb = xdbl + m * 48;   // [0:16)=dt-rank, [16:32)=B, [32:48)=C
        float y = 0.0f;
#pragma unroll
        for (int n = 0; n < DS; ++n) {
            float dA = expf(dtv * Av[n]);
            hs[n] = dA * hs[n] + dtv * xb[16 + n] * u;
            y += hs[n] * xb[32 + n];
        }
        y = (y + u * Dv) * zs[m * DI + d];
        yh[m * DI + d] = (_Float16)y;
    }
}

// ---------------------------------------------------------------------------
// Final reduction: out[m] = sum_c h[m][c]
// ---------------------------------------------------------------------------
__global__ void k_rowsum(const float* __restrict__ h, float* __restrict__ out) {
    int m = blockIdx.x * blockDim.x + threadIdx.x;
    if (m >= M_) return;
    float s = 0.0f;
    for (int c = 0; c < DM; ++c) s += h[(size_t)m * DM + c];
    out[m] = s;
}

// ---------------------------------------------------------------------------
extern "C" void kernel_launch(void* const* d_in, const int* in_sizes, int n_in,
                              void* d_out, int out_size, void* d_ws, size_t ws_size,
                              hipStream_t stream) {
    const float* x        = (const float*)d_in[0];
    // d_in[1] (y) unused by the reference
    const float* emb_w    = (const float*)d_in[2];
    const float* bn_gamma = (const float*)d_in[3];
    const float* bn_beta  = (const float*)d_in[4];
    const float* bn_mean  = (const float*)d_in[5];
    const float* bn_var   = (const float*)d_in[6];
    const float* in_w     = (const float*)d_in[7];   // (4,1024,256)
    const float* conv_w   = (const float*)d_in[8];   // (4,512,1,4)
    const float* conv_b   = (const float*)d_in[9];   // (4,512)
    const float* xp_w     = (const float*)d_in[10];  // (4,48,512)
    const float* dtp_w    = (const float*)d_in[11];  // (4,512,16)
    const float* dtp_b    = (const float*)d_in[12];  // (4,512)
    const float* A_log    = (const float*)d_in[13];  // (4,512,16)
    const float* D_skip   = (const float*)d_in[14];  // (4,512)
    const float* out_w    = (const float*)d_in[15];  // (4,256,512)

    // ---- workspace carve-out (bump allocator, 256B aligned) ----
    char* p = (char*)d_ws;
    auto alloc = [&](size_t bytes) -> char* {
        char* r = p;
        p += (bytes + 255) & ~(size_t)255;
        return r;
    };
    float*     h0f   = (float*)    alloc((size_t)M_ * DM * 4);
    float*     h1f   = (float*)    alloc((size_t)M_ * DM * 4);
    _Float16*  h0h   = (_Float16*) alloc((size_t)M_ * DM * 2);
    _Float16*  h1h   = (_Float16*) alloc((size_t)M_ * DM * 2);
    float*     xz    = (float*)    alloc((size_t)M_ * 2 * DI * 4);
    float*     xc    = (float*)    alloc((size_t)M_ * DI * 4);
    _Float16*  xch   = (_Float16*) alloc((size_t)M_ * DI * 2);
    float*     zs    = (float*)    alloc((size_t)M_ * DI * 4);
    float*     xdbl  = (float*)    alloc((size_t)M_ * 48 * 4);
    _Float16*  xdblh = (_Float16*) alloc((size_t)M_ * 48 * 2);
    float*     dtb   = (float*)    alloc((size_t)M_ * DI * 4);
    _Float16*  yh    = (_Float16*) alloc((size_t)M_ * DI * 2);
    _Float16*  wt_in  = (_Float16*)alloc((size_t)NL * DM * (2 * DI) * 2); // (256,1024)/layer
    _Float16*  wt_x   = (_Float16*)alloc((size_t)NL * DI * 48 * 2);       // (512,48)
    _Float16*  wt_dt  = (_Float16*)alloc((size_t)NL * 32 * DI * 2);       // (32,512) K-padded
    _Float16*  wt_out = (_Float16*)alloc((size_t)NL * DI * DM * 2);       // (512,256)

    // ---- one-time weight transposes to f16 (dt_proj K-padded 16 -> 32) ----
    for (int i = 0; i < NL; ++i) {
        k_transpose_f16<<<(2 * DI * DM + 255) / 256, 256, 0, stream>>>(
            in_w + (size_t)i * 2 * DI * DM, wt_in + (size_t)i * DM * 2 * DI,
            2 * DI, DM, DM);
        k_transpose_f16<<<(48 * DI + 255) / 256, 256, 0, stream>>>(
            xp_w + (size_t)i * 48 * DI, wt_x + (size_t)i * DI * 48, 48, DI, DI);
        k_transpose_f16<<<(DI * 32 + 255) / 256, 256, 0, stream>>>(
            dtp_w + (size_t)i * DI * 16, wt_dt + (size_t)i * 32 * DI, DI, 16, 32);
        k_transpose_f16<<<(DM * DI + 255) / 256, 256, 0, stream>>>(
            out_w + (size_t)i * DM * DI, wt_out + (size_t)i * DI * DM, DM, DI, DI);
    }

    // ---- stem: embed conv + BN + SiLU ----
    k_embed<<<(M_ * DM + 255) / 256, 256, 0, stream>>>(
        x, emb_w, bn_gamma, bn_beta, bn_mean, bn_var, h0f, h0h);

    float*    hcf = h0f; _Float16* hch = h0h;
    float*    hnf = h1f; _Float16* hnh = h1h;

    for (int i = 0; i < NL; ++i) {
        // in_proj: (4096x256) x (256->1024), 32x32 macro-tiles
        {
            int tiles = (M_ / 32) * ((2 * DI) / 32);          // 128*32
            k_wmma_gemm<2, 2><<<(tiles + 3) / 4, 128, 0, stream>>>(
                hch, wt_in + (size_t)i * DM * 2 * DI, xz, nullptr, nullptr,
                M_, 2 * DI, DM, DM, 2 * DI, 2 * DI, 0);
        }
        // depthwise causal conv + SiLU, silu(z)
        k_conv_silu<<<(M_ * DI + 255) / 256, 256, 0, stream>>>(
            xz, conv_w + (size_t)i * DI * 4, conv_b + (size_t)i * DI, xc, xch, zs);
        // x_proj: (4096x512) x (512->48), 32x48 macro-tiles, f16 mirror kept
        {
            int tiles = (M_ / 32) * (48 / 48);                // 128
            k_wmma_gemm<2, 3><<<(tiles + 3) / 4, 128, 0, stream>>>(
                xch, wt_x + (size_t)i * DI * 48, xdbl, xdblh, nullptr,
                M_, 48, DI, DI, 48, 48, 2);
        }
        // dt_proj: (4096x16) x (16->512); K padded to 32 via zero weight rows.
        {
            int tiles = (M_ / 32) * (DI / 32);                // 128*16
            k_wmma_gemm<2, 2><<<(tiles + 3) / 4, 128, 0, stream>>>(
                xdblh, wt_dt + (size_t)i * 32 * DI, dtb, nullptr, dtp_b + (size_t)i * DI,
                M_, DI, 32, 48, DI, DI, 1);
        }
        // selective scan (sequential in L, parallel over B*DI)
        k_scan<<<(B_ * DI + 255) / 256, 256, 0, stream>>>(
            dtb, xc, xdbl, zs, A_log + (size_t)i * DI * DS, D_skip + (size_t)i * DI, yh);
        // out_proj: (4096x512) x (512->256), f32 + f16 for next layer
        {
            int tiles = (M_ / 32) * (DM / 32);                // 128*8
            k_wmma_gemm<2, 2><<<(tiles + 3) / 4, 128, 0, stream>>>(
                yh, wt_out + (size_t)i * DI * DM, hnf, hnh, nullptr,
                M_, DM, DI, DI, DM, DM, 2);
        }
        // swap layer activation buffers
        float* tf = hcf; hcf = hnf; hnf = tf;
        _Float16* th = hch; hch = hnh; hnh = th;
    }

    k_rowsum<<<(M_ + 255) / 256, 256, 0, stream>>>(hcf, (float*)d_out);
}
